// CausalSelfAttention_18940805775481
// MI455X (gfx1250) — matrix operand order repaired and checked by reading, not compiled, not run on hardware
//
#include <hip/hip_runtime.h>
#include <hip/hip_bf16.h>

typedef __attribute__((ext_vector_type(16))) __bf16 v16bf;
typedef __attribute__((ext_vector_type(8)))  float  v8f;

#define D_EMBED 1024
#define N_HEADS 16
#define D_HEAD  64
#define BATCH   2
#define SEQ     2048

// ---- helpers -------------------------------------------------------------

// Starting K of bf16-pair p (0..7) inside a 16-bit A-fragment (16x32),
// contiguous order matching B frags: lanes0-15: V0..V7 -> K=0..15;
//                lanes16-31: V0..V7 -> K=16..31.
__device__ __forceinline__ int a_pair_k(int p, int hi) {
  return (hi ? 16 : 0) + 2 * p;
}

__device__ __forceinline__ v8f v8f_zero() {
  v8f z = {0.f, 0.f, 0.f, 0.f, 0.f, 0.f, 0.f, 0.f};
  return z;
}

__device__ __forceinline__ v8f wmma_bf16(v16bf a, v16bf b, v8f c) {
  // (neg_a, A, neg_b, B, c_mod, C, reuse_a, reuse_b)
  return __builtin_amdgcn_wmma_f32_16x16x32_bf16(false, a, false, b,
                                                 (short)0, c, false, false);
}

// Async global->LDS copy of 16 bytes (CDNA5 GLOBAL_LOAD_ASYNC_TO_LDS_B128,
// tracked on ASYNCcnt). lds_addr = LDS byte offset (low 32 bits of flat ptr).
__device__ __forceinline__ void async_copy_b128(unsigned lds_addr,
                                                unsigned long long gaddr) {
  asm volatile("global_load_async_to_lds_b128 %0, %1, off"
               :: "v"(lds_addr), "v"(gaddr) : "memory");
}
__device__ __forceinline__ void wait_asynccnt0() {
  asm volatile("s_wait_asynccnt 0" ::: "memory");
}

// ---- f32 -> bf16 conversion ---------------------------------------------

__global__ void cvt_kernel(const float* __restrict__ s, __bf16* __restrict__ d, int n) {
  int i = (blockIdx.x * blockDim.x + threadIdx.x) * 4;
  if (i + 3 < n) {
    float4 f = *(const float4*)(s + i);
    union { __bf16 h[4]; unsigned long long u; } U;
    U.h[0] = (__bf16)f.x; U.h[1] = (__bf16)f.y;
    U.h[2] = (__bf16)f.z; U.h[3] = (__bf16)f.w;
    *(unsigned long long*)(d + i) = U.u;
  } else {
    for (; i < n; ++i) d[i] = (__bf16)s[i];
  }
}

// ---- tiled WMMA GEMM: C[M x 1024] = A[M x 1024] * W[1024 x 1024] --------
// MODE 0: write bf16 in [B, H, S, 64] layout, scaled (QKV projections)
// MODE 1: write f32 row-major (output projection)

template <int MODE>
__global__ __launch_bounds__(128)
void gemm_wmma(const __bf16* __restrict__ A, const __bf16* __restrict__ W,
               __bf16* __restrict__ outb, float* __restrict__ outf, float scale)
{
  __shared__ __bf16 As[64][48];   // [m][k], padded row (96B, 16B-mult)
  __shared__ __bf16 Bt[64][40];   // [n][k] transposed, padded row (80B)

  const int tid = threadIdx.x;
  const int w  = tid >> 5;
  const int l  = tid & 31;
  const int lr = l & 15;
  const int hi = (l >> 4) & 1;
  const int bm = blockIdx.y * 64;
  const int bn = blockIdx.x * 64;

  v8f acc[4];
#pragma unroll
  for (int t = 0; t < 4; ++t) acc[t] = v8f_zero();

  for (int kk = 0; kk < D_EMBED; kk += 32) {
    { // A tile 64x32 -> LDS via async DMA (2 x b128 per thread)
      int r = tid >> 1, c = (tid & 1) * 16;
      unsigned long long g = (unsigned long long)(A + (size_t)(bm + r) * D_EMBED + kk + c);
      unsigned lo = (unsigned)(size_t)&As[r][c];
      async_copy_b128(lo, g);
      async_copy_b128(lo + 16, g + 16);
    }
    { // W tile 32x64 -> LDS transposed [n][k] (scatter, so stays on VGPR path)
      int r = tid >> 2, c = (tid & 3) * 16;
      union { uint4 u[2]; __bf16 h[16]; } U;
      const uint4* src = (const uint4*)(W + (size_t)(kk + r) * D_EMBED + bn + c);
      U.u[0] = src[0]; U.u[1] = src[1];
#pragma unroll
      for (int j = 0; j < 16; ++j) Bt[c + j][r] = U.h[j];
    }
    wait_asynccnt0();
    __syncthreads();

    v16bf a;
#pragma unroll
    for (int p = 0; p < 8; ++p)
      ((unsigned*)&a)[p] = *(const unsigned*)&As[w * 16 + lr][a_pair_k(p, hi)];

#pragma unroll
    for (int t = 0; t < 4; ++t) {
      v16bf b;  // B frag: elem j -> K=(hi?16:0)+j, N=t*16+lr  (contiguous in Bt)
      const __bf16* bp = &Bt[t * 16 + lr][hi * 16];
      ((uint4*)&b)[0] = *(const uint4*)bp;
      ((uint4*)&b)[1] = *(const uint4*)(bp + 8);
      acc[t] = wmma_bf16(a, b, acc[t]);
    }
    __syncthreads();
  }

#pragma unroll
  for (int t = 0; t < 4; ++t) {
#pragma unroll
    for (int r = 0; r < 8; ++r) {
      int row = bm + w * 16 + r + hi * 8;     // C layout: M = r + 8*hi
      int col = bn + t * 16 + lr;             // N = lane%16
      float val = acc[t][r] * scale;
      if constexpr (MODE == 0) {
        int b = row >> 11, s = row & (SEQ - 1);
        int h = col >> 6,  d = col & (D_HEAD - 1);
        outb[((size_t)(b * N_HEADS + h) * SEQ + s) * D_HEAD + d] = (__bf16)val;
      } else {
        outf[(size_t)row * D_EMBED + col] = val;
      }
    }
  }
}

// ---- flash attention: one (b,h) 64-row q-tile per block, 4 waves --------

__global__ __launch_bounds__(128)
void flash_attn(const __bf16* __restrict__ Q, const __bf16* __restrict__ K,
                const __bf16* __restrict__ V, __bf16* __restrict__ O)
{
  __shared__ __bf16 Vt[64][40];      // V tile transposed: [d][key_rel]
  __shared__ __bf16 Pl[4][16][40];   // per-wave P tile: [row][key_rel 0..31]

  const int tid = threadIdx.x;
  const int w  = tid >> 5;
  const int l  = tid & 31;
  const int lr = l & 15;
  const int hi = (l >> 4) & 1;
  const int qblk = blockIdx.x;       // 0..SEQ/64-1
  const int bh   = blockIdx.y;       // 0..B*H-1
  const size_t base = (size_t)bh * SEQ * D_HEAD;
  const __bf16* q = Q + base;
  const __bf16* k = K + base;
  const __bf16* v = V + base;

  const int qbase = qblk * 64 + w * 16;

  // Q A-fragments for d-chunks [0,32) and [32,64) (scale folded in at Q gemm)
  v16bf aq0, aq1;
#pragma unroll
  for (int p = 0; p < 8; ++p) {
    int kp = a_pair_k(p, hi);
    const __bf16* qp = q + (size_t)(qbase + lr) * D_HEAD;
    ((unsigned*)&aq0)[p] = *(const unsigned*)(qp + kp);
    ((unsigned*)&aq1)[p] = *(const unsigned*)(qp + 32 + kp);
  }

  v8f acc[4];
#pragma unroll
  for (int t = 0; t < 4; ++t) acc[t] = v8f_zero();
  float mrow[8], srow[8];
#pragma unroll
  for (int r = 0; r < 8; ++r) { mrow[r] = -3.0e38f; srow[r] = 0.0f; }

  const int kend = (qblk + 1) * 64;
  for (int kb = 0; kb < kend; kb += 32) {
    { // stage V[kb..kb+32) transposed into LDS (cooperative)
      int key = kb + (tid >> 2), c = (tid & 3) * 16;
      union { uint4 u[2]; __bf16 h[16]; } U;
      const uint4* src = (const uint4*)(v + (size_t)key * D_HEAD + c);
      U.u[0] = src[0]; U.u[1] = src[1];
#pragma unroll
      for (int j = 0; j < 16; ++j) Vt[c + j][tid >> 2] = U.h[j];
    }
    if (kb + 32 < kend) {  // prefetch next K/V tiles
      __builtin_prefetch(k + (size_t)(kb + 32 + lr) * D_HEAD, 0, 1);
      __builtin_prefetch(v + (size_t)(kb + 32 + lr) * D_HEAD, 0, 1);
    }
    __syncthreads();

    // scores: two 16x16 tiles, K-frags straight from global (contiguous in d)
    v8f s0 = v8f_zero(), s1 = v8f_zero();
    {
      const __bf16* kp = k + (size_t)(kb + lr) * D_HEAD + hi * 16;
      v16bf b;
      ((uint4*)&b)[0] = *(const uint4*)kp;        ((uint4*)&b)[1] = *(const uint4*)(kp + 8);
      s0 = wmma_bf16(aq0, b, s0);
      ((uint4*)&b)[0] = *(const uint4*)(kp + 32); ((uint4*)&b)[1] = *(const uint4*)(kp + 40);
      s0 = wmma_bf16(aq1, b, s0);
    }
    {
      const __bf16* kp = k + (size_t)(kb + 16 + lr) * D_HEAD + hi * 16;
      v16bf b;
      ((uint4*)&b)[0] = *(const uint4*)kp;        ((uint4*)&b)[1] = *(const uint4*)(kp + 8);
      s1 = wmma_bf16(aq0, b, s1);
      ((uint4*)&b)[0] = *(const uint4*)(kp + 32); ((uint4*)&b)[1] = *(const uint4*)(kp + 40);
      s1 = wmma_bf16(aq1, b, s1);
    }

    // causal mask + online softmax (row = r + 8*hi lives in a 16-lane group)
    const int key0 = kb + lr, key1 = kb + 16 + lr;
#pragma unroll
    for (int r = 0; r < 8; ++r) {
      const int qrow = qbase + r + hi * 8;
      float e0 = (key0 <= qrow) ? s0[r] : -3.0e38f;
      float e1 = (key1 <= qrow) ? s1[r] : -3.0e38f;
      float rm = fmaxf(e0, e1);
#pragma unroll
      for (int m = 1; m < 16; m <<= 1) rm = fmaxf(rm, __shfl_xor(rm, m, 32));
      float mnew = fmaxf(mrow[r], rm);
      float corr = __expf(mrow[r] - mnew);
      float p0 = __expf(e0 - mnew);
      float p1 = __expf(e1 - mnew);
      float rs = p0 + p1;
#pragma unroll
      for (int m = 1; m < 16; m <<= 1) rs += __shfl_xor(rs, m, 32);
      srow[r] = srow[r] * corr + rs;
      mrow[r] = mnew;
#pragma unroll
      for (int t = 0; t < 4; ++t) acc[t][r] *= corr;
      Pl[w][r + hi * 8][lr]      = (__bf16)p0;
      Pl[w][r + hi * 8][16 + lr] = (__bf16)p1;
    }
    // wave-private LDS RAW: DS ops are in-order per wave; stop compiler reordering
    asm volatile("" ::: "memory");

    // P (A-frag) from LDS, V (B-frags) from transposed LDS tile
    v16bf ap;
#pragma unroll
    for (int p = 0; p < 8; ++p)
      ((unsigned*)&ap)[p] = *(const unsigned*)&Pl[w][lr][a_pair_k(p, hi)];
#pragma unroll
    for (int t = 0; t < 4; ++t) {
      v16bf b;
      const __bf16* vp = &Vt[t * 16 + lr][hi * 16];
      ((uint4*)&b)[0] = *(const uint4*)vp;
      ((uint4*)&b)[1] = *(const uint4*)(vp + 8);
      acc[t] = wmma_bf16(ap, b, acc[t]);
    }
    __syncthreads();
  }

  // epilogue: normalize, write [B,S,E] bf16 for the output projection
  const int b = bh >> 4, h = bh & 15;
#pragma unroll
  for (int t = 0; t < 4; ++t) {
#pragma unroll
    for (int r = 0; r < 8; ++r) {
      int qrow = qbase + r + hi * 8;
      int d = t * 16 + lr;
      float val = acc[t][r] / srow[r];
      O[((size_t)b * SEQ + qrow) * D_EMBED + h * D_HEAD + d] = (__bf16)val;
    }
  }
}

// ---- launch --------------------------------------------------------------

extern "C" void kernel_launch(void* const* d_in, const int* in_sizes, int n_in,
                              void* d_out, int out_size, void* d_ws, size_t ws_size,
                              hipStream_t stream) {
  const float* x  = (const float*)d_in[0];
  const float* wq = (const float*)d_in[1];
  const float* wk = (const float*)d_in[2];
  const float* wv = (const float*)d_in[3];
  const float* wo = (const float*)d_in[4];
  float* out = (float*)d_out;

  const size_t NX = (size_t)BATCH * SEQ * D_EMBED;  // 4,194,304
  const size_t NW = (size_t)D_EMBED * D_EMBED;      // 1,048,576

  char* ws = (char*)d_ws;
  __bf16* xb  = (__bf16*)ws; ws += NX * 2;
  __bf16* wqb = (__bf16*)ws; ws += NW * 2;
  __bf16* wkb = (__bf16*)ws; ws += NW * 2;
  __bf16* wvb = (__bf16*)ws; ws += NW * 2;
  __bf16* wob = (__bf16*)ws; ws += NW * 2;
  __bf16* qb  = (__bf16*)ws; ws += NX * 2;
  __bf16* kb  = (__bf16*)ws; ws += NX * 2;
  __bf16* vb  = (__bf16*)ws; ws += NX * 2;
  __bf16* ab  = (__bf16*)ws; ws += NX * 2;

  cvt_kernel<<<(int)((NX / 4 + 255) / 256), 256, 0, stream>>>(x,  xb,  (int)NX);
  cvt_kernel<<<(int)((NW / 4 + 255) / 256), 256, 0, stream>>>(wq, wqb, (int)NW);
  cvt_kernel<<<(int)((NW / 4 + 255) / 256), 256, 0, stream>>>(wk, wkb, (int)NW);
  cvt_kernel<<<(int)((NW / 4 + 255) / 256), 256, 0, stream>>>(wv, wvb, (int)NW);
  cvt_kernel<<<(int)((NW / 4 + 255) / 256), 256, 0, stream>>>(wo, wob, (int)NW);

  dim3 ggrid(D_EMBED / 64, (BATCH * SEQ) / 64);     // (16, 64)
  gemm_wmma<0><<<ggrid, 128, 0, stream>>>(xb, wqb, qb, nullptr, 0.125f); // Q * 1/sqrt(64)
  gemm_wmma<0><<<ggrid, 128, 0, stream>>>(xb, wkb, kb, nullptr, 1.0f);
  gemm_wmma<0><<<ggrid, 128, 0, stream>>>(xb, wvb, vb, nullptr, 1.0f);

  dim3 agrid(SEQ / 64, BATCH * N_HEADS);            // (32, 32)
  flash_attn<<<agrid, 128, 0, stream>>>(qb, kb, vb, ab);

  gemm_wmma<1><<<ggrid, 128, 0, stream>>>(ab, wob, nullptr, out, 1.0f);
}
